// MultiHeadAttentionBlock_188978561586
// MI455X (gfx1250) — matrix operand order repaired and checked by reading, not compiled, not run on hardware
//
#include <hip/hip_runtime.h>
#include <math.h>

// Problem constants (match reference)
#define DM 1024          // d_model
#define NH 16            // heads
#define DK 64            // d_k
#define BB 4             // batch
#define SS 2048          // seq len
#define MM (BB * SS)     // total rows = 8192

typedef __attribute__((ext_vector_type(16))) _Float16 v16h;
typedef __attribute__((ext_vector_type(8)))  float    v8f;
typedef __attribute__((ext_vector_type(4)))  unsigned v4u;

union Frag {
    v16h     v;
    _Float16 h[16];
    unsigned u[8];
    v4u      q[2];
};

__device__ __forceinline__ v8f wmma_f16(v16h a, v16h b, v8f c) {
    // D = A(16x32 f16) * B(32x16 f16) + C(16x16 f32)
    return __builtin_amdgcn_wmma_f32_16x16x32_f16(
        /*neg_a=*/false, a, /*neg_b=*/false, b,
        /*c_mod=*/(short)0, c, /*reuse_a=*/false, /*reuse_b=*/false);
}

// A-fragment (16x32 f16): per lane, u[0..3] = K(half*8 .. +7) pairs,
// u[4..7] = K(16+half*8 ..): two 16-byte contiguous spans.
__device__ __forceinline__ void load_a_frag(Frag& f, const _Float16* base) {
    f.q[0] = *(const v4u*)(base);
    f.q[1] = *(const v4u*)(base + 16);
}
// B-fragment (32x16 f16): lane%16 = N column; h[s] = K(half*8+s) for s<8,
// K(16+half*8+s-8) for s>=8 (same K order as A), rows strided by ld.
__device__ __forceinline__ void load_b_frag(Frag& f, const _Float16* base, size_t ld) {
    const int hf = (threadIdx.x & 31) >> 4;
    #pragma unroll
    for (int s = 0; s < 16; ++s) {
        const int kk = (s < 8) ? (hf * 8 + s) : (16 + hf * 8 + (s - 8));
        f.h[s] = base[(size_t)kk * ld];
    }
}

// Reductions across the 16-lane half-groups (rows of the C layout)
__device__ __forceinline__ float rmax16(float v) {
    #pragma unroll
    for (int off = 1; off < 16; off <<= 1)
        v = fmaxf(v, __shfl_xor(v, off, 32));
    return v;
}
__device__ __forceinline__ float rsum16(float v) {
    #pragma unroll
    for (int off = 1; off < 16; off <<= 1)
        v += __shfl_xor(v, off, 32);
    return v;
}

// ---------------------------------------------------------------------------
// 1) fp32 -> fp16 elementwise convert (4 elements / thread)
// ---------------------------------------------------------------------------
__global__ void cvt_f32_to_f16(const float* __restrict__ x,
                               _Float16* __restrict__ y, size_t n4) {
    size_t i = ((size_t)blockIdx.x * blockDim.x + threadIdx.x);
    if (i >= n4) return;
    size_t idx = i * 4;
    float4 f = *(const float4*)(x + idx);
    y[idx + 0] = (_Float16)f.x;
    y[idx + 1] = (_Float16)f.y;
    y[idx + 2] = (_Float16)f.z;
    y[idx + 3] = (_Float16)f.w;
}

// ---------------------------------------------------------------------------
// 2) mask (int32) -> additive f32 bias: 0 where allowed, -1e9 where masked.
//    Through softmax this is equivalent to the reference's where(mask==0,-1e9).
// ---------------------------------------------------------------------------
__global__ void mask_to_bias(const int* __restrict__ mask,
                             float* __restrict__ bias, size_t n4) {
    size_t i = ((size_t)blockIdx.x * blockDim.x + threadIdx.x);
    if (i >= n4) return;
    size_t idx = i * 4;
    int4 m = *(const int4*)(mask + idx);
    float4 o;
    o.x = (m.x == 0) ? -1e9f : 0.f;
    o.y = (m.y == 0) ? -1e9f : 0.f;
    o.z = (m.z == 0) ? -1e9f : 0.f;
    o.w = (m.w == 0) ? -1e9f : 0.f;
    *(float4*)(bias + idx) = o;
}

// ---------------------------------------------------------------------------
// 3) W[N,K] fp32 -> Wt[K,N] fp16 (transpose so GEMM B-frag lanes read adjacent N)
// ---------------------------------------------------------------------------
__global__ void cvt_transpose_f16(const float* __restrict__ W,
                                  _Float16* __restrict__ Wt, int N, int K) {
    int k = blockIdx.x * 32 + (threadIdx.x & 31);
    int n = blockIdx.y * 8 + (threadIdx.x >> 5);
    if (k < K && n < N)
        Wt[(size_t)k * N + n] = (_Float16)W[(size_t)n * K + k];
}

// ---------------------------------------------------------------------------
// 4) GEMM: Out = A[M,K](f16) * Bt[K,N](f16) + bias[N](f32)
//    MODE 0: f32 out [M,N]; MODE 1: f16 out [M,N];
//    MODE 2: f16 out, K-projection transposed layout Kt[B,H,DK,S]
//    Block = 256 thr = 8 waves tiled 4(M) x 2(N).
//    Each wave: 32(M) x 64(N) = 2x4 WMMA accumulators -> block tile 128x128.
//    Per K-step: 4 A-loads + 4 B-frag gathers (16 halves each) feed 8 WMMAs (B frags reused 2x).
// ---------------------------------------------------------------------------
template <int MODE>
__global__ __launch_bounds__(256)
void gemm_bias(const _Float16* __restrict__ A, const _Float16* __restrict__ Bt,
               const float* __restrict__ bias, void* __restrict__ Out,
               int M, int N, int K) {
    const int lane = threadIdx.x & 31;
    const int wave = threadIdx.x >> 5;           // 0..7
    const int wm   = wave & 3;                   // 0..3  (M tiles)
    const int wn   = wave >> 2;                  // 0..1  (N tiles)
    const int m0   = blockIdx.y * 128 + wm * 32;
    const int n0   = blockIdx.x * 128 + wn * 64;
    const int mloc = lane & 15;
    const int half = lane >> 4;

    const v8f vzero = {0.f, 0.f, 0.f, 0.f, 0.f, 0.f, 0.f, 0.f};
    v8f acc[2][4];
    #pragma unroll
    for (int mt = 0; mt < 2; ++mt)
        #pragma unroll
        for (int j = 0; j < 4; ++j) acc[mt][j] = vzero;

    const _Float16* arow0 = A + (size_t)(m0 + mloc) * K + half * 8;
    const _Float16* arow1 = arow0 + (size_t)16 * K;

    for (int k0 = 0; k0 < K; k0 += 32) {
        __builtin_prefetch((const void*)(arow0 + k0 + 128), 0, 0);
        __builtin_prefetch((const void*)(arow1 + k0 + 128), 0, 0);
        Frag a[2];
        load_a_frag(a[0], arow0 + k0);
        load_a_frag(a[1], arow1 + k0);
        const _Float16* brow = Bt + (size_t)k0 * N + n0 + mloc;
        #pragma unroll
        for (int j = 0; j < 4; ++j) {
            Frag b;
            load_b_frag(b, brow + j * 16, (size_t)N);
            acc[0][j] = wmma_f16(a[0].v, b.v, acc[0][j]);
            acc[1][j] = wmma_f16(a[1].v, b.v, acc[1][j]);
        }
    }

    // C layout: VGPR r, lanes 0-15 -> (M=r, N=lane); lanes 16-31 -> (M=r+8)
    #pragma unroll
    for (int mt = 0; mt < 2; ++mt)
        #pragma unroll
        for (int j = 0; j < 4; ++j) {
            const int n  = n0 + j * 16 + mloc;
            const float bv = bias[n];
            #pragma unroll
            for (int r = 0; r < 8; ++r) {
                const int m   = m0 + mt * 16 + r + half * 8;
                const float val = acc[mt][j][r] + bv;
                if (MODE == 0) {
                    ((float*)Out)[(size_t)m * N + n] = val;
                } else if (MODE == 1) {
                    ((_Float16*)Out)[(size_t)m * N + n] = (_Float16)val;
                } else {
                    // Kt[b, h, d, s]: n = h*64 + d; m = b*SS + s
                    const int h = n >> 6, d = n & 63;
                    const int bidx = m >> 11, s = m & (SS - 1);
                    ((_Float16*)Out)[(((size_t)bidx * NH + h) * DK + d) * SS + s] =
                        (_Float16)val;
                }
            }
        }
}

// ---------------------------------------------------------------------------
// 5) Flash attention.  Qp/Vp layout [B,S, h*64+d] (f16); Kt layout [B,H,DK,S].
//    Block = 128 thr = 4 waves; each wave owns 32 query rows (2 Q-tiles) of
//    one (b,h): K/V fragments are reused by both Q-tiles.
//    Mask pre-folded into additive f32 bias (one FMA per score element).
//    Online softmax; P tiles re-swizzled C->A layout through LDS (b128 ds ops).
// ---------------------------------------------------------------------------
__global__ __launch_bounds__(128)
void flash_attn(const _Float16* __restrict__ Qp, const _Float16* __restrict__ Kt,
                const _Float16* __restrict__ Vp, const float* __restrict__ sbias,
                _Float16* __restrict__ ctx) {
    __shared__ __align__(16) _Float16 lds_p[4][2 * 16 * 32];   // per wave, 2 P tiles

    const int lane = threadIdx.x & 31;
    const int wave = threadIdx.x >> 5;
    const int b    = blockIdx.z;
    const int h    = blockIdx.y;
    const int q0   = blockIdx.x * 128 + wave * 32;   // 32 query rows per wave
    const int mloc = lane & 15;
    const int half = lane >> 4;

    const _Float16* Qb  = Qp + ((size_t)b * SS + q0) * DM + h * DK;
    const _Float16* Ktb = Kt + ((size_t)b * NH + h) * DK * SS;   // [d][s]
    const _Float16* Vb  = Vp + (size_t)b * SS * DM + h * DK;
    const float* biasb  = sbias + (size_t)b * SS * SS;

    // Q fragments: [qt][c], c = d-chunk (0..31 / 32..63)
    Frag qa[2][2];
    #pragma unroll
    for (int qt = 0; qt < 2; ++qt)
        #pragma unroll
        for (int c = 0; c < 2; ++c)
            load_a_frag(qa[qt][c],
                        Qb + (size_t)(qt * 16 + mloc) * DM + c * 32 + half * 8);

    const v8f vzero = {0.f, 0.f, 0.f, 0.f, 0.f, 0.f, 0.f, 0.f};
    v8f acc[2][4];
    #pragma unroll
    for (int qt = 0; qt < 2; ++qt)
        #pragma unroll
        for (int j = 0; j < 4; ++j) acc[qt][j] = vzero;

    float row_m[2][8], row_l[2][8];
    #pragma unroll
    for (int qt = 0; qt < 2; ++qt)
        #pragma unroll
        for (int r = 0; r < 8; ++r) { row_m[qt][r] = -1e30f; row_l[qt][r] = 0.f; }

    const float scale = 0.125f;  // 1/sqrt(64)

    for (int kt = 0; kt < SS; kt += 32) {
        // ---- K fragments for this key block: [t = key sub-tile][c = d chunk]
        Frag kb[2][2];
        #pragma unroll
        for (int t = 0; t < 2; ++t)
            #pragma unroll
            for (int c = 0; c < 2; ++c)
                load_b_frag(kb[t][c],
                            Ktb + (size_t)(c * 32) * SS + kt + t * 16 + mloc, (size_t)SS);

        #pragma unroll
        for (int qt = 0; qt < 2; ++qt) {
            // ---- scores: two 16x16 C tiles covering keys kt..kt+31 ----
            v8f s[2];
            #pragma unroll
            for (int t = 0; t < 2; ++t) {
                v8f cacc = vzero;
                #pragma unroll
                for (int c = 0; c < 2; ++c)
                    cacc = wmma_f16(qa[qt][c].v, kb[t][c].v, cacc);
                s[t] = cacc;
            }

            // ---- scale + additive mask bias + online softmax update ----
            #pragma unroll
            for (int r = 0; r < 8; ++r) {
                const int qrow = q0 + qt * 16 + r + half * 8;
                const float* brow = biasb + (size_t)qrow * SS + kt + mloc;
                const float v0 = fmaf(s[0][r], scale, brow[0]);
                const float v1 = fmaf(s[1][r], scale, brow[16]);
                const float tmax  = rmax16(fmaxf(v0, v1));
                const float mnew  = fmaxf(row_m[qt][r], tmax);
                const float alpha = __expf(row_m[qt][r] - mnew);
                const float e0 = __expf(v0 - mnew);
                const float e1 = __expf(v1 - mnew);
                row_l[qt][r] = row_l[qt][r] * alpha + rsum16(e0 + e1);
                row_m[qt][r] = mnew;
                #pragma unroll
                for (int j = 0; j < 4; ++j) acc[qt][j][r] *= alpha;

                // stage P elements (C layout) into LDS
                const int m = r + 8 * half;
                lds_p[wave][qt * 512 + m * 32 + mloc]      = (_Float16)e0;
                lds_p[wave][qt * 512 + m * 32 + 16 + mloc] = (_Float16)e1;
            }
        }
        __syncthreads();   // uniform trip count across all 4 waves

        // ---- reload P in A layout (2x ds_load_b128 per tile) ----
        Frag pa[2];
        #pragma unroll
        for (int qt = 0; qt < 2; ++qt)
            load_a_frag(pa[qt], &lds_p[wave][qt * 512 + mloc * 32 + half * 8]);
        __syncthreads();

        // ---- ctx += P(16x32) * V(32x64): V frags reused by both Q-tiles ----
        const _Float16* vrow = Vb + (size_t)kt * DM + mloc;
        #pragma unroll
        for (int j = 0; j < 4; ++j) {
            Frag vb;   // B(n=d -> lane%16, k=key over elements): strided gather
            load_b_frag(vb, vrow + j * 16, (size_t)DM);
            acc[0][j] = wmma_f16(pa[0].v, vb.v, acc[0][j]);
            acc[1][j] = wmma_f16(pa[1].v, vb.v, acc[1][j]);
        }
    }

    // ---- finalize: ctx / l, store f16 at [b, s, h*64 + d] ----
    #pragma unroll
    for (int qt = 0; qt < 2; ++qt)
        #pragma unroll
        for (int j = 0; j < 4; ++j)
            #pragma unroll
            for (int r = 0; r < 8; ++r) {
                const int m = q0 + qt * 16 + r + half * 8;
                const float val = acc[qt][j][r] / row_l[qt][r];
                ctx[((size_t)b * SS + m) * DM + h * DK + j * 16 + mloc] =
                    (_Float16)val;
            }
}

// ---------------------------------------------------------------------------
// Host-side orchestration
// ---------------------------------------------------------------------------
extern "C" void kernel_launch(void* const* d_in, const int* in_sizes, int n_in,
                              void* d_out, int out_size, void* d_ws, size_t ws_size,
                              hipStream_t stream) {
    const float* q    = (const float*)d_in[0];
    const float* k    = (const float*)d_in[1];
    const float* v    = (const float*)d_in[2];
    const int*   mask = (const int*)  d_in[3];
    const float* Wq   = (const float*)d_in[4];
    const float* bq   = (const float*)d_in[5];
    const float* Wk   = (const float*)d_in[6];
    const float* bk   = (const float*)d_in[7];
    const float* Wv   = (const float*)d_in[8];
    const float* bv   = (const float*)d_in[9];
    const float* Wo   = (const float*)d_in[10];
    const float* bo   = (const float*)d_in[11];
    float* out = (float*)d_out;

    const size_t szX = (size_t)MM * DM;   // 8192*1024 elements
    const size_t szW = (size_t)DM * DM;
    const size_t szM = (size_t)BB * SS * SS;   // mask/bias elements

    _Float16* p = (_Float16*)d_ws;
    _Float16* q16  = p; p += szX;
    _Float16* k16  = p; p += szX;
    _Float16* v16  = p; p += szX;
    _Float16* WqT  = p; p += szW;
    _Float16* WkT  = p; p += szW;
    _Float16* WvT  = p; p += szW;
    _Float16* WoT  = p; p += szW;
    _Float16* Qp   = p; p += szX;
    _Float16* Ktp  = p; p += szX;   // [B,H,DK,S]
    _Float16* Vp   = p; p += szX;
    _Float16* ctx  = p; p += szX;
    float*    sbias = (float*)p;    // [B,S,S] additive mask bias (f32)

    // 1) convert activations to f16; expand mask to additive bias
    {
        const size_t n4 = szX / 4;
        const int blk = 256;
        const int grd = (int)((n4 + blk - 1) / blk);
        cvt_f32_to_f16<<<grd, blk, 0, stream>>>(q, q16, n4);
        cvt_f32_to_f16<<<grd, blk, 0, stream>>>(k, k16, n4);
        cvt_f32_to_f16<<<grd, blk, 0, stream>>>(v, v16, n4);

        const size_t m4 = szM / 4;
        const int grdm = (int)((m4 + blk - 1) / blk);
        mask_to_bias<<<grdm, blk, 0, stream>>>(mask, sbias, m4);
    }

    // 2) convert + transpose weights: W[N,K] -> Wt[K,N]
    {
        dim3 grd(DM / 32, DM / 8);
        cvt_transpose_f16<<<grd, 256, 0, stream>>>(Wq, WqT, DM, DM);
        cvt_transpose_f16<<<grd, 256, 0, stream>>>(Wk, WkT, DM, DM);
        cvt_transpose_f16<<<grd, 256, 0, stream>>>(Wv, WvT, DM, DM);
        cvt_transpose_f16<<<grd, 256, 0, stream>>>(Wo, WoT, DM, DM);
    }

    // 3) Q/K/V projections (f16 out; K written pre-transposed per head)
    {
        dim3 grd(DM / 128, MM / 128);
        gemm_bias<1><<<grd, 256, 0, stream>>>(q16, WqT, bq, (void*)Qp,  MM, DM, DM);
        gemm_bias<2><<<grd, 256, 0, stream>>>(k16, WkT, bk, (void*)Ktp, MM, DM, DM);
        gemm_bias<1><<<grd, 256, 0, stream>>>(v16, WvT, bv, (void*)Vp,  MM, DM, DM);
    }

    // 4) attention: 128 query rows per block (32 per wave)
    {
        dim3 grd(SS / 128, NH, BB);
        flash_attn<<<grd, 128, 0, stream>>>(Qp, Ktp, Vp, sbias, ctx);
    }

    // 5) output projection (f32 out)
    {
        dim3 grd(DM / 128, MM / 128);
        gemm_bias<0><<<grd, 256, 0, stream>>>(ctx, WoT, bo, (void*)out, MM, DM, DM);
    }
}
